// PairTripletLoss_49546742727248
// MI455X (gfx1250) — matrix operand order repaired and checked by reading, not compiled, not run on hardware
//
#include <hip/hip_runtime.h>

#define NN 4096
#define DD 256
#define BM 128
#define BN 128
#define KS 32
#define SP 40      // padded LDS row stride in ushorts (80B rows: 16B-aligned, conflict-free)
#define MT 4       // 16x16 tiles per wave in M
#define NT 2       // 16x16 tiles per wave in N
#define TILE_USH (BM * SP)       // ushorts per LDS tile (5120 -> 10240 B)
#define NTILES 6                 // Ah, Al, B1h, B1l, B2h, B2l
#define STAGE_USH (NTILES * TILE_USH)
#define MARGIN_F 0.5f

typedef __attribute__((ext_vector_type(16))) __bf16 v16bf;
typedef __attribute__((ext_vector_type(8)))  float v8f;

union Frag { v16bf v; uint4 q[2]; };
union Acc  { v8f  v; float f[8]; };

__device__ __forceinline__ unsigned short f2bf(float f) {
  unsigned u = __float_as_uint(f);
  u += 0x7FFFu + ((u >> 16) & 1u);   // round-to-nearest-even
  return (unsigned short)(u >> 16);
}
__device__ __forceinline__ float bf2f(unsigned short h) {
  return __uint_as_float(((unsigned)h) << 16);
}

// CDNA5 async direct copy: global memory -> LDS, 16 bytes per lane, ASYNCcnt-tracked.
// VDST = 32-bit LDS byte address (low 32 bits of the flat LDS pointer), VADDR = 64-bit
// global address, SADDR = off.  (ISA 15.18.3, opcode 98)
__device__ __forceinline__ void async_copy_b128(const void* gptr, void* lptr) {
  unsigned lds = (unsigned)(uintptr_t)lptr;
  unsigned long long ga = (unsigned long long)(uintptr_t)gptr;
  asm volatile("global_load_async_to_lds_b128 %0, %1, off"
               :: "v"(lds), "v"(ga) : "memory");
}
__device__ __forceinline__ void wait_async0() {
  asm volatile("s_wait_asynccnt 0x0" ::: "memory");
}

// One wave32 per row: norms, dist_ap, and bf16 hi/lo split of both matrices.
__global__ __launch_bounds__(256)
void triplet_prep_kernel(const float* __restrict__ p1, const float* __restrict__ p2,
                         unsigned short* __restrict__ p1h, unsigned short* __restrict__ p1l,
                         unsigned short* __restrict__ p2h, unsigned short* __restrict__ p2l,
                         float* __restrict__ n1, float* __restrict__ n2,
                         float* __restrict__ dap) {
  const int wave = threadIdx.x >> 5;
  const int lane = threadIdx.x & 31;
  const int row  = blockIdx.x * 8 + wave;
  const float* r1 = p1 + (size_t)row * DD;
  const float* r2 = p2 + (size_t)row * DD;
  float s1 = 0.f, s2 = 0.f, sd = 0.f;
  for (int k = lane; k < DD; k += 32) {
    float a = r1[k], b = r2[k];
    s1 += a * a; s2 += b * b;
    float d = a - b; sd += d * d;
    unsigned short ah = f2bf(a);
    unsigned short al = f2bf(a - bf2f(ah));
    unsigned short bh = f2bf(b);
    unsigned short bl = f2bf(b - bf2f(bh));
    size_t idx = (size_t)row * DD + k;
    p1h[idx] = ah; p1l[idx] = al;
    p2h[idx] = bh; p2l[idx] = bl;
  }
  #pragma unroll
  for (int off = 16; off > 0; off >>= 1) {
    s1 += __shfl_down(s1, off, 32);
    s2 += __shfl_down(s2, off, 32);
    sd += __shfl_down(sd, off, 32);
  }
  if (lane == 0) { n1[row] = s1; n2[row] = s2; dap[row] = sd; }
}

// 128x128 tile per block; 8 waves (2x4); each wave owns 4x2 16x16 WMMA tiles for BOTH
// C1 = P1·P1^T and C2 = P1·P2^T, via bf16 hi/lo split (3 WMMAs per product).
// Double-buffered LDS filled by async-to-LDS copies; one s_wait_asynccnt + barrier / kstep.
__global__ __launch_bounds__(256)
void triplet_gemm_hinge_kernel(const unsigned short* __restrict__ p1h,
                               const unsigned short* __restrict__ p1l,
                               const unsigned short* __restrict__ p2h,
                               const unsigned short* __restrict__ p2l,
                               const float* __restrict__ n1,
                               const float* __restrict__ n2,
                               const float* __restrict__ dap,
                               float* __restrict__ partials) {
  __shared__ unsigned short smem[2 * STAGE_USH];   // 2 x 60 KB
  __shared__ float red[256];

  const int tid   = threadIdx.x;
  const int lane  = tid & 31;
  const int wave  = tid >> 5;
  const int waveM = wave >> 2;        // 0..1 -> 64-row stripe
  const int waveN = wave & 3;         // 0..3 -> 32-col stripe
  const int blockRow = blockIdx.y * BM;
  const int blockCol = blockIdx.x * BN;

  Acc acc1[MT][NT], acc2[MT][NT];
  #pragma unroll
  for (int mt = 0; mt < MT; ++mt)
    #pragma unroll
    for (int nt = 0; nt < NT; ++nt)
      #pragma unroll
      for (int e = 0; e < 8; ++e) { acc1[mt][nt].f[e] = 0.f; acc2[mt][nt].f[e] = 0.f; }

  const int l15 = lane & 15;
  const int akb = (lane >> 4) * 16;   // A frag K-base: 0 or 16
  const int bkb = (lane >> 4) * 16;   // B frag K-base: 0 or 16

  // thread's fixed staging slot: 2 chunks of 16B per tile per kstep
  const int lin0 = tid;               // chunk 0: rows 0..63
  const int lin1 = tid + 256;         // chunk 1: rows 64..127
  const int row0 = lin0 >> 2, c0 = (lin0 & 3) * 8;
  const int row1 = lin1 >> 2, c1 = (lin1 & 3) * 8;

  // ---- async stage of one kstep into buffer `st` ----
  auto stage = [&](int st, int kk) {
    unsigned short* b = smem + st * STAGE_USH;
    size_t gA0 = (size_t)(blockRow + row0) * DD + kk + c0;
    size_t gA1 = (size_t)(blockRow + row1) * DD + kk + c1;
    size_t gB0 = (size_t)(blockCol + row0) * DD + kk + c0;
    size_t gB1 = (size_t)(blockCol + row1) * DD + kk + c1;
    int d0 = row0 * SP + c0, d1 = row1 * SP + c1;
    async_copy_b128(p1h + gA0, b + 0 * TILE_USH + d0);
    async_copy_b128(p1h + gA1, b + 0 * TILE_USH + d1);
    async_copy_b128(p1l + gA0, b + 1 * TILE_USH + d0);
    async_copy_b128(p1l + gA1, b + 1 * TILE_USH + d1);
    async_copy_b128(p1h + gB0, b + 2 * TILE_USH + d0);
    async_copy_b128(p1h + gB1, b + 2 * TILE_USH + d1);
    async_copy_b128(p1l + gB0, b + 3 * TILE_USH + d0);
    async_copy_b128(p1l + gB1, b + 3 * TILE_USH + d1);
    async_copy_b128(p2h + gB0, b + 4 * TILE_USH + d0);
    async_copy_b128(p2h + gB1, b + 4 * TILE_USH + d1);
    async_copy_b128(p2l + gB0, b + 5 * TILE_USH + d0);
    async_copy_b128(p2l + gB1, b + 5 * TILE_USH + d1);
  };

  // prologue: fill buffer 0
  stage(0, 0);
  wait_async0();
  __syncthreads();

  for (int kk = 0; kk < DD; kk += KS) {
    const int cur = (kk / KS) & 1;
    if (kk + KS < DD) stage(cur ^ 1, kk + KS);   // prefetch next kstep (no wait)

    const unsigned short* sb  = smem + cur * STAGE_USH;
    const unsigned short* Ahs = sb + 0 * TILE_USH;
    const unsigned short* Als = sb + 1 * TILE_USH;
    const unsigned short* B1h = sb + 2 * TILE_USH;
    const unsigned short* B1l = sb + 3 * TILE_USH;
    const unsigned short* B2h = sb + 4 * TILE_USH;
    const unsigned short* B2l = sb + 5 * TILE_USH;

    // B fragments are mt-invariant: load once per kstep
    Frag fB1h[NT], fB1l[NT], fB2h[NT], fB2l[NT];
    #pragma unroll
    for (int nt = 0; nt < NT; ++nt) {
      const int bo = (waveN * 32 + nt * 16 + l15) * SP + bkb;
      fB1h[nt].q[0] = *(const uint4*)(B1h + bo);
      fB1h[nt].q[1] = *(const uint4*)(B1h + bo + 8);
      fB1l[nt].q[0] = *(const uint4*)(B1l + bo);
      fB1l[nt].q[1] = *(const uint4*)(B1l + bo + 8);
      fB2h[nt].q[0] = *(const uint4*)(B2h + bo);
      fB2h[nt].q[1] = *(const uint4*)(B2h + bo + 8);
      fB2l[nt].q[0] = *(const uint4*)(B2l + bo);
      fB2l[nt].q[1] = *(const uint4*)(B2l + bo + 8);
    }

    #pragma unroll
    for (int mt = 0; mt < MT; ++mt) {
      const int ao = (waveM * 64 + mt * 16 + l15) * SP + akb;
      Frag fAh, fAl;
      fAh.q[0] = *(const uint4*)(Ahs + ao);
      fAh.q[1] = *(const uint4*)(Ahs + ao + 8);
      fAl.q[0] = *(const uint4*)(Als + ao);
      fAl.q[1] = *(const uint4*)(Als + ao + 8);
      #pragma unroll
      for (int nt = 0; nt < NT; ++nt) {
        // C1 = P1·P1^T : hi*hi + hi*lo + lo*hi
        acc1[mt][nt].v = __builtin_amdgcn_wmma_f32_16x16x32_bf16(
            false, fAh.v, false, fB1h[nt].v, (short)0, acc1[mt][nt].v, false, false);
        acc1[mt][nt].v = __builtin_amdgcn_wmma_f32_16x16x32_bf16(
            false, fAh.v, false, fB1l[nt].v, (short)0, acc1[mt][nt].v, false, false);
        acc1[mt][nt].v = __builtin_amdgcn_wmma_f32_16x16x32_bf16(
            false, fAl.v, false, fB1h[nt].v, (short)0, acc1[mt][nt].v, false, false);
        // C2 = P1·P2^T
        acc2[mt][nt].v = __builtin_amdgcn_wmma_f32_16x16x32_bf16(
            false, fAh.v, false, fB2h[nt].v, (short)0, acc2[mt][nt].v, false, false);
        acc2[mt][nt].v = __builtin_amdgcn_wmma_f32_16x16x32_bf16(
            false, fAh.v, false, fB2l[nt].v, (short)0, acc2[mt][nt].v, false, false);
        acc2[mt][nt].v = __builtin_amdgcn_wmma_f32_16x16x32_bf16(
            false, fAl.v, false, fB2h[nt].v, (short)0, acc2[mt][nt].v, false, false);
      }
    }

    wait_async0();       // next-stage fills complete (this wave)
    __syncthreads();     // everyone done computing `cur` + staging `cur^1`
  }

  // ---- epilogue: distances + hinge, masked diagonal ----
  // C layout: lanes 0-15 -> N=lane, M=r; lanes 16-31 -> N=lane-16, M=r+8
  float lsum = 0.f;
  #pragma unroll
  for (int mt = 0; mt < MT; ++mt) {
    const int i0 = blockRow + waveM * 64 + mt * 16 + 8 * (lane >> 4);
    float ni[8], di[8];
    #pragma unroll
    for (int r = 0; r < 8; ++r) { ni[r] = n1[i0 + r]; di[r] = dap[i0 + r]; }
    #pragma unroll
    for (int nt = 0; nt < NT; ++nt) {
      const int j = blockCol + waveN * 32 + nt * 16 + l15;
      const float n1j = n1[j];
      const float n2j = n2[j];
      #pragma unroll
      for (int r = 0; r < 8; ++r) {
        const int i = i0 + r;
        float d11 = ni[r] + n1j - 2.0f * acc1[mt][nt].f[r];
        float d12 = ni[r] + n2j - 2.0f * acc2[mt][nt].f[r];
        float h = fmaxf(di[r] - d11 + MARGIN_F, 0.f)
                + fmaxf(di[r] - d12 + MARGIN_F, 0.f);
        lsum += (i != j) ? h : 0.f;
      }
    }
  }

  // deterministic block reduction
  red[tid] = lsum;
  __syncthreads();
  #pragma unroll
  for (int s = 128; s > 0; s >>= 1) {
    if (tid < s) red[tid] += red[tid + s];
    __syncthreads();
  }
  if (tid == 0) partials[blockIdx.y * gridDim.x + blockIdx.x] = red[0];
}

__global__ __launch_bounds__(256)
void triplet_finalize_kernel(const float* __restrict__ partials, int np,
                             float* __restrict__ out) {
  __shared__ float red[256];
  float s = 0.f;
  for (int i = threadIdx.x; i < np; i += 256) s += partials[i];
  red[threadIdx.x] = s;
  __syncthreads();
  #pragma unroll
  for (int k = 128; k > 0; k >>= 1) {
    if (threadIdx.x < k) red[threadIdx.x] += red[threadIdx.x + k];
    __syncthreads();
  }
  if (threadIdx.x == 0) {
    double cnt = (double)NN * (double)(NN - 1);
    out[0] = (float)((double)red[0] / cnt);
  }
}

extern "C" void kernel_launch(void* const* d_in, const int* in_sizes, int n_in,
                              void* d_out, int out_size, void* d_ws, size_t ws_size,
                              hipStream_t stream) {
  (void)in_sizes; (void)n_in; (void)out_size; (void)ws_size;
  const float* p1 = (const float*)d_in[0];
  const float* p2 = (const float*)d_in[1];

  char* ws = (char*)d_ws;
  size_t off = 0;
  const size_t plane = (size_t)NN * DD * sizeof(unsigned short);  // 2 MB
  unsigned short* p1h = (unsigned short*)(ws + off); off += plane;
  unsigned short* p1l = (unsigned short*)(ws + off); off += plane;
  unsigned short* p2h = (unsigned short*)(ws + off); off += plane;
  unsigned short* p2l = (unsigned short*)(ws + off); off += plane;
  float* n1  = (float*)(ws + off); off += (size_t)NN * sizeof(float);
  float* n2  = (float*)(ws + off); off += (size_t)NN * sizeof(float);
  float* dap = (float*)(ws + off); off += (size_t)NN * sizeof(float);
  float* partials = (float*)(ws + off);

  triplet_prep_kernel<<<NN / 8, 256, 0, stream>>>(p1, p2, p1h, p1l, p2h, p2l, n1, n2, dap);

  dim3 grid(NN / BN, NN / BM);   // 32 x 32 = 1024 blocks
  triplet_gemm_hinge_kernel<<<grid, 256, 0, stream>>>(p1h, p1l, p2h, p2l,
                                                      n1, n2, dap, partials);

  triplet_finalize_kernel<<<1, 256, 0, stream>>>(partials, (NN / BM) * (NN / BN),
                                                 (float*)d_out);
}
